// EnhancedGCN_4200478015989
// MI455X (gfx1250) — compile-verified, matrix operand order checked by reading, not run on hardware
//
#include <hip/hip_runtime.h>
#include <math.h>

#define HD    256
#define HEADS 4
#define DH    64
#define NG    64

typedef _Float16 f16;
typedef __attribute__((ext_vector_type(16))) _Float16 v16h;
typedef __attribute__((ext_vector_type(2)))  _Float16 v2h;
typedef __attribute__((ext_vector_type(8)))  float    v8f;

__device__ __forceinline__ float lrelu(float x, float s) { return x > 0.f ? x : x * s; }

__device__ __forceinline__ void atomicMaxF(float* addr, float val) {
  if (val >= 0.f) atomicMax((int*)addr, __float_as_int(val));
  else            atomicMin((unsigned int*)addr, __float_as_uint(val));
}

// ---------------------------------------------------------------------------
// WMMA GEMM: C[M,Ncols] = act(A[M,K](f32->f16) @ B[K,Ncols](f32->f16) + bias)
// block = 256 threads (8 waves), tile 128x64, wave tile 16x64 (4 wmma accs)
// mode 0: none   mode 1: leaky(slope)   mode 2: BN(leaky(.,0.2))
// ---------------------------------------------------------------------------
#define TILE_M 128
#define TILE_N 64
#define TILE_K 32
#define LDSP   40   // padded row pitch (halves) to skew LDS banks

__global__ __launch_bounds__(256) void wmma_gemm_f32(
    const float* __restrict__ A, int lda, int acol,
    const float* __restrict__ B, int ldb,
    float* __restrict__ C, int ldc, int ccol,
    const float* __restrict__ bias,
    const float* __restrict__ bn_g, const float* __restrict__ bn_b,
    const float* __restrict__ bn_m, const float* __restrict__ bn_v,
    int M, int K, int mode, float slope)
{
  __shared__ f16 Alds[TILE_M][LDSP];
  __shared__ f16 Blds[TILE_N][LDSP];   // B stored transposed: [col][k]

  const int tid  = threadIdx.x;
  const int lane = tid & 31;
  const int wave = tid >> 5;
  const int khalf = lane >> 4;   // 0: K low group, 1: K high group per ISA layout
  const int l16   = lane & 15;
  const int rowBase = blockIdx.x * TILE_M;
  const int colBase = blockIdx.y * TILE_N;

  v8f acc[4] = {};

  for (int k0 = 0; k0 < K; k0 += TILE_K) {
    __syncthreads();
    // A tile 128x32 (guard M tail with zeros)
    for (int i = tid; i < TILE_M * TILE_K; i += 256) {
      int r = i >> 5, kk = i & 31;
      int gr = rowBase + r;
      float v = (gr < M) ? A[(long)gr * lda + acol + k0 + kk] : 0.f;
      Alds[r][kk] = (f16)v;
    }
    // B tile 32x64, transposed into LDS
    for (int i = tid; i < TILE_N * TILE_K; i += 256) {
      int c = i & 63, kk = i >> 6;
      Blds[c][kk] = (f16)B[(long)(k0 + kk) * ldb + colBase + c];
    }
    __syncthreads();

    // A fragment: lane = row M (mod 16); VGPR j holds K pair per ISA 16-bit layout
    v16h af;
#pragma unroll
    for (int j = 0; j < 8; ++j) {
      int kk = ((j & 4) << 2) + khalf * 8 + ((j & 3) << 1);
      v2h p = *(const v2h*)&Alds[wave * 16 + l16][kk];
      af[2 * j] = p.x; af[2 * j + 1] = p.y;
    }
    // Build ALL four B fragments first so the DS loads batch into one clause,
    // then chain the four WMMAs back-to-back (XDL co-execution friendly).
    v16h bf[4];
#pragma unroll
    for (int t = 0; t < 4; ++t) {
#pragma unroll
      for (int j = 0; j < 8; ++j) {
        int kk = ((j & 4) << 2) + khalf * 8 + ((j & 3) << 1);
        v2h p = *(const v2h*)&Blds[t * 16 + l16][kk];
        bf[t][2 * j] = p.x; bf[t][2 * j + 1] = p.y;
      }
    }
#pragma unroll
    for (int t = 0; t < 4; ++t) {
      acc[t] = __builtin_amdgcn_wmma_f32_16x16x32_f16(
          false, af, false, bf[t], (short)0, acc[t], false, false);
    }
  }

  // C layout: lane%16 = N col, (lane/16)*8 + vgpr = M row
#pragma unroll
  for (int t = 0; t < 4; ++t) {
#pragma unroll
    for (int r = 0; r < 8; ++r) {
      int m = khalf * 8 + r;
      int grow = rowBase + wave * 16 + m;
      if (grow >= M) continue;
      int gc = colBase + t * 16 + l16;     // column index into bias/bn
      float v = acc[t][r];
      if (bias) v += bias[gc];
      if (mode == 1) v = lrelu(v, slope);
      else if (mode == 2) {
        v = lrelu(v, 0.2f);
        v = (v - bn_m[gc]) * rsqrtf(bn_v[gc] + 1e-5f) * bn_g[gc] + bn_b[gc];
      }
      C[(long)grow * ldc + ccol + gc] = v;
    }
  }
}

// ---------------------------------------------------------------------------
// Elementwise / scatter kernels
// ---------------------------------------------------------------------------
__global__ void fill_kernel(float* p, float v, long n) {
  long i = (long)blockIdx.x * blockDim.x + threadIdx.x;
  if (i < n) p[i] = v;
}

__global__ void fuse_c_kernel(const float* __restrict__ x, const float* __restrict__ Wc,
                              const float* __restrict__ bc,
                              const float* __restrict__ g, const float* __restrict__ b,
                              const float* __restrict__ m, const float* __restrict__ v,
                              float* __restrict__ ce, int N) {
  int i = blockIdx.x * blockDim.x + threadIdx.x;
  if (i >= N * 32) return;
  int n = i >> 5, j = i & 31;
  float acc = bc[j];
#pragma unroll
  for (int k = 0; k < 7; ++k) acc += x[(long)n * 135 + k] * Wc[k * 32 + j];
  acc = lrelu(acc, 0.2f);
  acc = (acc - m[j]) * rsqrtf(v[j] + 1e-5f) * g[j] + b[j];
  ce[(long)n * 96 + j] = acc;
}

__global__ void deg_acc(const int* __restrict__ ei, float* deg, int E) {
  int e = blockIdx.x * blockDim.x + threadIdx.x;
  if (e < E) atomicAdd(&deg[ei[E + e]], 1.f);   // dst
}
__global__ void dis_kernel(const float* __restrict__ deg, float* dis, int N) {
  int i = blockIdx.x * blockDim.x + threadIdx.x;
  if (i < N) { float d = deg[i]; dis[i] = d > 0.f ? rsqrtf(d) : 0.f; }
}
__global__ void norm_kernel(const int* __restrict__ ei, const float* __restrict__ dis,
                            float* nrm, int E) {
  int e = blockIdx.x * blockDim.x + threadIdx.x;
  if (e < E) nrm[e] = dis[ei[e]] * dis[ei[E + e]];
}

__global__ void rowbias_init(float* __restrict__ out, const float* __restrict__ bias, int N) {
  long i = (long)blockIdx.x * blockDim.x + threadIdx.x;
  if (i < (long)N * HD) out[i] = bias[i & (HD - 1)];
}

// one block per (edge or self-loop), 256 lanes = 256 channels: coalesced atomics
__global__ void gcn_scatter(const int* __restrict__ ei, const float* __restrict__ nrm,
                            const float* __restrict__ dis, const float* __restrict__ hh,
                            float* __restrict__ out, int E, int N) {
  int e = blockIdx.x, c = threadIdx.x;
  int s, d; float w;
  if (e < E) { s = ei[e]; d = ei[E + e]; w = nrm[e]; }
  else       { s = d = e - E; float di = dis[s]; w = di * di; }
  atomicAdd(&out[(long)d * HD + c], hh[(long)s * HD + c] * w);
}

__global__ void att_coef(const float* __restrict__ g, const float* __restrict__ asrc,
                         const float* __restrict__ adst,
                         float* __restrict__ a_s, float* __restrict__ a_d, int N) {
  int i = blockIdx.x * blockDim.x + threadIdx.x;
  if (i >= N * HEADS) return;
  int n = i >> 2, hd = i & 3;
  const float* gp = g + (long)n * HD + hd * DH;
  float s = 0.f, d = 0.f;
#pragma unroll 8
  for (int k = 0; k < DH; ++k) { s += gp[k] * asrc[hd * DH + k]; d += gp[k] * adst[hd * DH + k]; }
  a_s[i] = s; a_d[i] = d;
}

__global__ void att_init(float* mbuf, float* den, int N) {
  int i = blockIdx.x * blockDim.x + threadIdx.x;
  if (i < N * HEADS) { mbuf[i] = -INFINITY; den[i] = 0.f; }
}

__global__ void att_max(const int* __restrict__ ei, const float* __restrict__ a_s,
                        const float* __restrict__ a_d, float* __restrict__ mbuf,
                        int E, int N) {
  int i = blockIdx.x * blockDim.x + threadIdx.x;
  if (i >= (E + N) * HEADS) return;
  int e = i >> 2, hd = i & 3;
  int s, d;
  if (e < E) { s = ei[e]; d = ei[E + e]; } else { s = d = e - E; }
  float logit = lrelu(a_s[s * HEADS + hd] + a_d[d * HEADS + hd], 0.2f);
  atomicMaxF(&mbuf[d * HEADS + hd], logit);
}

__global__ void att_den(const int* __restrict__ ei, const float* __restrict__ a_s,
                        const float* __restrict__ a_d, const float* __restrict__ mbuf,
                        float* __restrict__ den, int E, int N) {
  int i = blockIdx.x * blockDim.x + threadIdx.x;
  if (i >= (E + N) * HEADS) return;
  int e = i >> 2, hd = i & 3;
  int s, d;
  if (e < E) { s = ei[e]; d = ei[E + e]; } else { s = d = e - E; }
  float logit = lrelu(a_s[s * HEADS + hd] + a_d[d * HEADS + hd], 0.2f);
  atomicAdd(&den[d * HEADS + hd], __expf(logit - mbuf[d * HEADS + hd]));
}

__global__ void gat_scatter(const int* __restrict__ ei, const float* __restrict__ a_s,
                            const float* __restrict__ a_d, const float* __restrict__ mbuf,
                            const float* __restrict__ den, const float* __restrict__ g,
                            float* __restrict__ out, int E, int N) {
  int e = blockIdx.x, c = threadIdx.x;
  int hd = c >> 6;
  int s, d;
  if (e < E) { s = ei[e]; d = ei[E + e]; } else { s = d = e - E; }
  float logit = lrelu(a_s[s * HEADS + hd] + a_d[d * HEADS + hd], 0.2f);
  float alpha = __expf(logit - mbuf[d * HEADS + hd]) / den[d * HEADS + hd];
  atomicAdd(&out[(long)d * HD + c], g[(long)s * HD + c] * alpha);
}

__global__ void gat_post(const float* __restrict__ out,
                         const float* __restrict__ bg, const float* __restrict__ bb,
                         const float* __restrict__ bm, const float* __restrict__ bv,
                         float* __restrict__ h, float* __restrict__ maxbuf, int N) {
  long i = (long)blockIdx.x * blockDim.x + threadIdx.x;
  if (i >= (long)N * HD) return;
  int c = i & (HD - 1);
  float v = out[i];
  v = (v - bm[c]) * rsqrtf(bv[c] + 1e-5f) * bg[c] + bb[c];
  v = lrelu(v, 0.2f);
  h[i] = v;
  maxbuf[i] = fmaxf(maxbuf[i], v);
}

__global__ void pool_acc(const float* __restrict__ h1, const int* __restrict__ batch,
                         float* __restrict__ gsum, int N) {
  long i = (long)blockIdx.x * blockDim.x + threadIdx.x;
  if (i >= (long)N * HD) return;
  int n = (int)(i >> 8), c = (int)(i & 255);
  atomicAdd(&gsum[batch[n] * HD + c], h1[i]);
}
__global__ void pool_add(float* __restrict__ h1, const int* __restrict__ batch,
                         const float* __restrict__ gsum, int N) {
  long i = (long)blockIdx.x * blockDim.x + threadIdx.x;
  if (i >= (long)N * HD) return;
  int n = (int)(i >> 8), c = (int)(i & 255);
  h1[i] += 0.1f * gsum[batch[n] * HD + c];
}

__global__ void score_kernel(const float* __restrict__ z, const float* __restrict__ w2,
                             const float* __restrict__ b2, const float* __restrict__ temp,
                             float* __restrict__ t, int N) {
  int n = blockIdx.x * blockDim.x + threadIdx.x;
  if (n >= N) return;
  float acc = b2[0];
#pragma unroll 8
  for (int k = 0; k < HD; ++k) acc += z[(long)n * HD + k] * w2[k];
  float s = 1.f / (1.f + __expf(-acc));
  t[n] = s / temp[0];
}

__global__ void red_init(float* scal) { scal[0] = -INFINITY; scal[1] = 0.f; }

__global__ void red_max(const float* __restrict__ t, float* scal, int N) {
  __shared__ float sm[256];
  int n = blockIdx.x * 256 + threadIdx.x;
  sm[threadIdx.x] = (n < N) ? t[n] : -INFINITY;
  __syncthreads();
  for (int s = 128; s > 0; s >>= 1) {
    if ((int)threadIdx.x < s) sm[threadIdx.x] = fmaxf(sm[threadIdx.x], sm[threadIdx.x + s]);
    __syncthreads();
  }
  if (threadIdx.x == 0) atomicMaxF(&scal[0], sm[0]);
}
__global__ void red_sum(const float* __restrict__ t, float* scal, int N) {
  __shared__ float sm[256];
  int n = blockIdx.x * 256 + threadIdx.x;
  sm[threadIdx.x] = (n < N) ? __expf(t[n] - scal[0]) : 0.f;
  __syncthreads();
  for (int s = 128; s > 0; s >>= 1) {
    if ((int)threadIdx.x < s) sm[threadIdx.x] += sm[threadIdx.x + s];
    __syncthreads();
  }
  if (threadIdx.x == 0) atomicAdd(&scal[1], sm[0]);
}
__global__ void final_kernel(const float* __restrict__ t, const float* __restrict__ scal,
                             float* __restrict__ out, int N) {
  int n = blockIdx.x * blockDim.x + threadIdx.x;
  if (n >= N) return;
  float ex = __expf(t[n] - scal[0]) / scal[1];
  out[n] = ex * ex;
}

// ---------------------------------------------------------------------------
extern "C" void kernel_launch(void* const* d_in, const int* in_sizes, int n_in,
                              void* d_out, int out_size, void* d_ws, size_t ws_size,
                              hipStream_t stream) {
  const float* x     = (const float*)d_in[0];
  const int*   ei    = (const int*)d_in[1];
  const int*   batch = (const int*)d_in[2];
  const float* Wc = (const float*)d_in[3],  *bc = (const float*)d_in[4];
  const float* bnc_g = (const float*)d_in[5], *bnc_b = (const float*)d_in[6];
  const float* bnc_m = (const float*)d_in[7], *bnc_v = (const float*)d_in[8];
  const float* We = (const float*)d_in[9],  *be = (const float*)d_in[10];
  const float* bne_g = (const float*)d_in[11], *bne_b = (const float*)d_in[12];
  const float* bne_m = (const float*)d_in[13], *bne_v = (const float*)d_in[14];
  const float* Wf = (const float*)d_in[15], *bf = (const float*)d_in[16];
  const float* gcnW0 = (const float*)d_in[17];
  const float* gcnW  = (const float*)d_in[18];
  const float* gcnb  = (const float*)d_in[19];
  const float* gatW  = (const float*)d_in[20];
  const float* asrc  = (const float*)d_in[21];
  const float* adst  = (const float*)d_in[22];
  const float* gatb  = (const float*)d_in[23];
  const float* bng = (const float*)d_in[24], *bnb = (const float*)d_in[25];
  const float* bnm = (const float*)d_in[26], *bnv = (const float*)d_in[27];
  const float* lin1W = (const float*)d_in[28], *lin1b = (const float*)d_in[29];
  const float* mlpW1 = (const float*)d_in[30], *mlpb1 = (const float*)d_in[31];
  const float* mlpW2 = (const float*)d_in[32], *mlpb2 = (const float*)d_in[33];
  const float* temp  = (const float*)d_in[34];

  const int N = in_sizes[2];
  const int E = in_sizes[1] / 2;

  // workspace bump allocator
  char* w = (char*)d_ws;
  auto alloc = [&](size_t bytes) -> float* {
    float* p = (float*)w;
    w += (bytes + 255) & ~(size_t)255;
    return p;
  };
  float* ce     = alloc((size_t)N * 96 * 4);
  float* h      = alloc((size_t)N * HD * 4);
  float* hh     = alloc((size_t)N * HD * 4);   // also GAT out / lin1 out
  float* g      = alloc((size_t)N * HD * 4);   // also mlp z
  float* maxbuf = alloc((size_t)N * HD * 4);
  float* deg    = alloc((size_t)N * 4);
  float* dis    = alloc((size_t)N * 4);
  float* nrm    = alloc((size_t)E * 4);
  float* a_s    = alloc((size_t)N * HEADS * 4);
  float* a_d    = alloc((size_t)N * HEADS * 4);
  float* mbuf   = alloc((size_t)N * HEADS * 4);
  float* den    = alloc((size_t)N * HEADS * 4);
  float* gsum   = alloc((size_t)NG * HD * 4);
  float* tsc    = alloc((size_t)N * 4);
  float* scal   = alloc(2 * 4);

  auto gemm = [&](const float* A, int lda, int acol, const float* B, int Ncols,
                  float* C, int ldc, int ccol, const float* bias,
                  const float* g0, const float* b0, const float* m0, const float* v0,
                  int M, int K, int mode, float slope) {
    dim3 grid((M + TILE_M - 1) / TILE_M, Ncols / TILE_N);
    wmma_gemm_f32<<<grid, dim3(256), 0, stream>>>(A, lda, acol, B, Ncols, C, ldc, ccol,
                                                  bias, g0, b0, m0, v0, M, K, mode, slope);
  };

  const long NHD = (long)N * HD;
  const int  BLK = 256;

  // ---- FeatureFusion ----
  fuse_c_kernel<<<(N * 32 + BLK - 1) / BLK, BLK, 0, stream>>>(
      x, Wc, bc, bnc_g, bnc_b, bnc_m, bnc_v, ce, N);
  // e = BN(lrelu(x[:,7:] @ We + be)) -> ce[:,32:96]
  gemm(x, 135, 7, We, 64, ce, 96, 32, be, bne_g, bne_b, bne_m, bne_v, N, 128, 2, 0.2f);
  // h = ce @ Wf + bf  (stored [N,64])
  gemm(ce, 96, 0, Wf, 64, h, 64, 0, bf, nullptr, nullptr, nullptr, nullptr, N, 96, 0, 0.f);

  // ---- GCN symmetric norm (self loops included) ----
  fill_kernel<<<(N + BLK - 1) / BLK, BLK, 0, stream>>>(deg, 1.f, N);  // self-loop contributes 1
  deg_acc<<<(E + BLK - 1) / BLK, BLK, 0, stream>>>(ei, deg, E);
  dis_kernel<<<(N + BLK - 1) / BLK, BLK, 0, stream>>>(deg, dis, N);
  norm_kernel<<<(E + BLK - 1) / BLK, BLK, 0, stream>>>(ei, dis, nrm, E);

  fill_kernel<<<(unsigned)((NHD + BLK - 1) / BLK), BLK, 0, stream>>>(maxbuf, -INFINITY, NHD);

  // ---- 4 GCN+GAT layers ----
  for (int i = 0; i < 4; ++i) {
    const float* W = (i == 0) ? gcnW0 : gcnW + (size_t)(i - 1) * HD * HD;
    const int Kd = (i == 0) ? 64 : HD;
    // hh = h @ W
    gemm(h, Kd, 0, W, HD, hh, HD, 0,
         nullptr, nullptr, nullptr, nullptr, nullptr, N, Kd, 0, 0.f);
    // h = gcnb[i] (+ scatter)
    rowbias_init<<<(unsigned)((NHD + BLK - 1) / BLK), BLK, 0, stream>>>(h, gcnb + i * HD, N);
    gcn_scatter<<<E + N, HD, 0, stream>>>(ei, nrm, dis, hh, h, E, N);
    // g = h @ gatW[i]
    gemm(h, HD, 0, gatW + (size_t)i * HD * HD, HD, g, HD, 0,
         nullptr, nullptr, nullptr, nullptr, nullptr, N, HD, 0, 0.f);
    att_coef<<<(N * HEADS + BLK - 1) / BLK, BLK, 0, stream>>>(
        g, asrc + i * HEADS * DH, adst + i * HEADS * DH, a_s, a_d, N);
    att_init<<<(N * HEADS + BLK - 1) / BLK, BLK, 0, stream>>>(mbuf, den, N);
    att_max<<<((E + N) * HEADS + BLK - 1) / BLK, BLK, 0, stream>>>(ei, a_s, a_d, mbuf, E, N);
    att_den<<<((E + N) * HEADS + BLK - 1) / BLK, BLK, 0, stream>>>(ei, a_s, a_d, mbuf, den, E, N);
    // hh = gatb[i] (+ attention-weighted scatter of g)
    rowbias_init<<<(unsigned)((NHD + BLK - 1) / BLK), BLK, 0, stream>>>(hh, gatb + i * HD, N);
    gat_scatter<<<E + N, HD, 0, stream>>>(ei, a_s, a_d, mbuf, den, g, hh, E, N);
    // h = lrelu(BN(hh)); maxbuf = max(maxbuf, h)
    gat_post<<<(unsigned)((NHD + BLK - 1) / BLK), BLK, 0, stream>>>(
        hh, bng + i * HD, bnb + i * HD, bnm + i * HD, bnv + i * HD, h, maxbuf, N);
  }

  // ---- lin1 on JK-max, pooling, MLP ----
  gemm(maxbuf, HD, 0, lin1W, HD, hh, HD, 0, lin1b,
       nullptr, nullptr, nullptr, nullptr, N, HD, 1, 0.01f);
  fill_kernel<<<(NG * HD + BLK - 1) / BLK, BLK, 0, stream>>>(gsum, 0.f, (long)NG * HD);
  pool_acc<<<(unsigned)((NHD + BLK - 1) / BLK), BLK, 0, stream>>>(hh, batch, gsum, N);
  pool_add<<<(unsigned)((NHD + BLK - 1) / BLK), BLK, 0, stream>>>(hh, batch, gsum, N);
  gemm(hh, HD, 0, mlpW1, HD, g, HD, 0, mlpb1,
       nullptr, nullptr, nullptr, nullptr, N, HD, 1, 0.2f);
  score_kernel<<<(N + BLK - 1) / BLK, BLK, 0, stream>>>(g, mlpW2, mlpb2, temp, tsc, N);

  // ---- global softmax + square ----
  red_init<<<1, 1, 0, stream>>>(scal);
  red_max<<<(N + BLK - 1) / BLK, BLK, 0, stream>>>(tsc, scal, N);
  red_sum<<<(N + BLK - 1) / BLK, BLK, 0, stream>>>(tsc, scal, N);
  final_kernel<<<(N + BLK - 1) / BLK, BLK, 0, stream>>>(tsc, scal, (float*)d_out, N);
}
